// SparseDeformableChannelMambaBlock_22016002359947
// MI455X (gfx1250) — compile-verified, matrix operand order checked by reading, not run on hardware
//
#include <hip/hip_runtime.h>
#include <hip/hip_bf16.h>

#define B_    8
#define L_    2048
#define D_    256
#define EXP_  512
#define NST_  16
#define DCONV_ 4
#define KTOP_ 614   // int(2048 * 0.3)

typedef _Float16 v16h __attribute__((ext_vector_type(16)));
typedef _Float16 h8   __attribute__((ext_vector_type(8)));
typedef float    v8f  __attribute__((ext_vector_type(8)));

// ---------------- f32 -> f16 convert ----------------
__global__ void k_cvt_f16(const float* __restrict__ s, _Float16* __restrict__ d, int n) {
    int i = blockIdx.x * blockDim.x + threadIdx.x;
    if (i < n) d[i] = (_Float16)s[i];
}

// ---------------- DyT: tanh(alpha*x)*w + b, emit f16 ----------------
__global__ void k_dyt(const float* __restrict__ x, const float* __restrict__ alpha,
                      const float* __restrict__ w, const float* __restrict__ b,
                      _Float16* __restrict__ out, int n) {
    int i = blockIdx.x * blockDim.x + threadIdx.x;
    if (i >= n) return;
    int d = i & (D_ - 1);
    float v = tanhf(alpha[0] * x[i]) * w[d] + b[d];
    out[i] = (_Float16)v;
}

__device__ __forceinline__ v16h load_frag_f16(const _Float16* __restrict__ p, int k0, int k1) {
    h8 lo = *(const h8*)(p + k0);
    h8 hi = *(const h8*)(p + k1);
    v16h f;
#pragma unroll
    for (int j = 0; j < 8; ++j) { f[j] = lo[j]; f[j + 8] = hi[j]; }
    return f;
}

__device__ __forceinline__ v16h load_frag_f32(const float* __restrict__ p, int k0, int k1) {
    v16h f;
#pragma unroll
    for (int j = 0; j < 8; ++j) {
        f[j]     = (_Float16)p[k0 + j];
        f[j + 8] = (_Float16)p[k1 + j];
    }
    return f;
}

// ---------------- 2x2 register-blocked WMMA GEMM ----------------
// A: f16 MxK row-major, B: f16 NxK row-major  =>  C = A * B^T + bias (f32 MxN)
// Each wave produces a 32x32 output block (4 accumulators); A/B fragments reused 2x each.
__global__ void k_gemm2x2_f16(const _Float16* __restrict__ A, const _Float16* __restrict__ Bm,
                              const float* __restrict__ bias, float* __restrict__ C,
                              int M, int N, int K) {
    int wavesPerBlock = blockDim.x >> 5;
    int wave = blockIdx.x * wavesPerBlock + (threadIdx.x >> 5);
    int lane = threadIdx.x & 31;
    int tilesN2 = N >> 5;
    int tilesM2 = (M + 31) >> 5;
    if (wave >= tilesM2 * tilesN2) return;      // wave-uniform
    int tm2 = wave / tilesN2;
    int tn2 = wave - tm2 * tilesN2;
    int half = lane >> 4, r = lane & 15;
    int ra0 = tm2 * 32 + r;
    int ra1 = ra0 + 16;
    if (ra0 > M - 1) ra0 = M - 1;               // clamp loads; results discarded by store guard
    if (ra1 > M - 1) ra1 = M - 1;
    const _Float16* Ar0 = A  + (size_t)ra0 * K;
    const _Float16* Ar1 = A  + (size_t)ra1 * K;
    const _Float16* Br0 = Bm + (size_t)(tn2 * 32 + r) * K;
    const _Float16* Br1 = Bm + (size_t)(tn2 * 32 + 16 + r) * K;
    v8f c00 = {}, c01 = {}, c10 = {}, c11 = {};
    for (int kk = 0; kk < K; kk += 32) {
        int k0 = kk + half * 8;
        int k1 = kk + 16 + half * 8;
        v16h a0 = load_frag_f16(Ar0, k0, k1);
        v16h a1 = load_frag_f16(Ar1, k0, k1);
        v16h b0 = load_frag_f16(Br0, k0, k1);
        v16h b1 = load_frag_f16(Br1, k0, k1);
        c00 = __builtin_amdgcn_wmma_f32_16x16x32_f16(false, a0, false, b0, (short)0, c00, false, false);
        c01 = __builtin_amdgcn_wmma_f32_16x16x32_f16(false, a0, false, b1, (short)0, c01, false, false);
        c10 = __builtin_amdgcn_wmma_f32_16x16x32_f16(false, a1, false, b0, (short)0, c10, false, false);
        c11 = __builtin_amdgcn_wmma_f32_16x16x32_f16(false, a1, false, b1, (short)0, c11, false, false);
    }
    int col0 = tn2 * 32 + r;
    int col1 = col0 + 16;
    float bz0 = bias[col0], bz1 = bias[col1];
#pragma unroll
    for (int v = 0; v < 8; ++v) {
        int row0 = tm2 * 32 + v + half * 8;
        int row1 = row0 + 16;
        if (row0 < M) {
            C[(size_t)row0 * N + col0] = c00[v] + bz0;
            C[(size_t)row0 * N + col1] = c01[v] + bz1;
        }
        if (row1 < M) {
            C[(size_t)row1 * N + col0] = c10[v] + bz0;
            C[(size_t)row1 * N + col1] = c11[v] + bz1;
        }
    }
}

// Same, but A is f32 (converted to f16 on load).
__global__ void k_gemm2x2_a32(const float* __restrict__ A, const _Float16* __restrict__ Bm,
                              const float* __restrict__ bias, float* __restrict__ C,
                              int M, int N, int K) {
    int wavesPerBlock = blockDim.x >> 5;
    int wave = blockIdx.x * wavesPerBlock + (threadIdx.x >> 5);
    int lane = threadIdx.x & 31;
    int tilesN2 = N >> 5;
    int tilesM2 = (M + 31) >> 5;
    if (wave >= tilesM2 * tilesN2) return;
    int tm2 = wave / tilesN2;
    int tn2 = wave - tm2 * tilesN2;
    int half = lane >> 4, r = lane & 15;
    int ra0 = tm2 * 32 + r;
    int ra1 = ra0 + 16;
    if (ra0 > M - 1) ra0 = M - 1;
    if (ra1 > M - 1) ra1 = M - 1;
    const float*    Ar0 = A  + (size_t)ra0 * K;
    const float*    Ar1 = A  + (size_t)ra1 * K;
    const _Float16* Br0 = Bm + (size_t)(tn2 * 32 + r) * K;
    const _Float16* Br1 = Bm + (size_t)(tn2 * 32 + 16 + r) * K;
    v8f c00 = {}, c01 = {}, c10 = {}, c11 = {};
    for (int kk = 0; kk < K; kk += 32) {
        int k0 = kk + half * 8;
        int k1 = kk + 16 + half * 8;
        v16h a0 = load_frag_f32(Ar0, k0, k1);
        v16h a1 = load_frag_f32(Ar1, k0, k1);
        v16h b0 = load_frag_f16(Br0, k0, k1);
        v16h b1 = load_frag_f16(Br1, k0, k1);
        c00 = __builtin_amdgcn_wmma_f32_16x16x32_f16(false, a0, false, b0, (short)0, c00, false, false);
        c01 = __builtin_amdgcn_wmma_f32_16x16x32_f16(false, a0, false, b1, (short)0, c01, false, false);
        c10 = __builtin_amdgcn_wmma_f32_16x16x32_f16(false, a1, false, b0, (short)0, c10, false, false);
        c11 = __builtin_amdgcn_wmma_f32_16x16x32_f16(false, a1, false, b1, (short)0, c11, false, false);
    }
    int col0 = tn2 * 32 + r;
    int col1 = col0 + 16;
    float bz0 = bias[col0], bz1 = bias[col1];
#pragma unroll
    for (int v = 0; v < 8; ++v) {
        int row0 = tm2 * 32 + v + half * 8;
        int row1 = row0 + 16;
        if (row0 < M) {
            C[(size_t)row0 * N + col0] = c00[v] + bz0;
            C[(size_t)row0 * N + col1] = c01[v] + bz1;
        }
        if (row1 < M) {
            C[(size_t)row1 * N + col0] = c10[v] + bz0;
            C[(size_t)row1 * N + col1] = c11[v] + bz1;
        }
    }
}

// ---------------- row L2-normalize (in place) + Q/K scalars ----------------
__global__ void k_norm_qk(float* __restrict__ xp,
                          const float* __restrict__ wq, const float* __restrict__ bq,
                          const float* __restrict__ wk, const float* __restrict__ bk,
                          float* __restrict__ Qv, float* __restrict__ Kv) {
    int row = blockIdx.x;
    float* p = xp + (size_t)row * EXP_;
    float ss = 0.f, qs = 0.f, ks = 0.f;
    for (int e = threadIdx.x; e < EXP_; e += blockDim.x) {
        float v = p[e];
        ss += v * v; qs += v * wq[e]; ks += v * wk[e];
    }
#pragma unroll
    for (int off = 16; off; off >>= 1) {
        ss += __shfl_xor(ss, off, 32);
        qs += __shfl_xor(qs, off, 32);
        ks += __shfl_xor(ks, off, 32);
    }
    __shared__ float sss[32], sqs[32], sks[32];
    int wid = threadIdx.x >> 5, nw = blockDim.x >> 5;
    if ((threadIdx.x & 31) == 0) { sss[wid] = ss; sqs[wid] = qs; sks[wid] = ks; }
    __syncthreads();
    if (threadIdx.x == 0) {
        float S = 0, Q = 0, Kk = 0;
        for (int i = 0; i < nw; ++i) { S += sss[i]; Q += sqs[i]; Kk += sks[i]; }
        sss[0] = S; sqs[0] = Q; sks[0] = Kk;
    }
    __syncthreads();
    float nrm = fmaxf(sqrtf(sss[0]), 1e-12f);
    float inv = 1.0f / nrm;
    for (int e = threadIdx.x; e < EXP_; e += blockDim.x) p[e] *= inv;
    if (threadIdx.x == 0) {
        Qv[row] = sqs[0] * inv + bq[0];
        Kv[row] = sks[0] * inv + bk[0];
    }
}

// ---------------- rank-1 attention importance: imp[m] = mean_l softmax_m(Q_l*K_m) ----------------
__global__ void __launch_bounds__(1024) k_imp(const float* __restrict__ Qv, const float* __restrict__ Kv,
                                              float* __restrict__ imp) {
    __shared__ float Ks[L_], Qs[L_], Ss[L_], Rs[L_];
    __shared__ float smax[32], smin[32];
    int b = blockIdx.x, t = threadIdx.x;
    for (int i = t; i < L_; i += blockDim.x) { Ks[i] = Kv[b * L_ + i]; Qs[i] = Qv[b * L_ + i]; }
    __syncthreads();
    float mx = -1e30f, mn = 1e30f;
    for (int i = t; i < L_; i += blockDim.x) { mx = fmaxf(mx, Ks[i]); mn = fminf(mn, Ks[i]); }
#pragma unroll
    for (int off = 16; off; off >>= 1) {
        mx = fmaxf(mx, __shfl_xor(mx, off, 32));
        mn = fminf(mn, __shfl_xor(mn, off, 32));
    }
    if ((t & 31) == 0) { smax[t >> 5] = mx; smin[t >> 5] = mn; }
    __syncthreads();
    if (t == 0) {
        float M = -1e30f, m = 1e30f;
        for (int i = 0; i < (int)(blockDim.x >> 5); ++i) { M = fmaxf(M, smax[i]); m = fminf(m, smin[i]); }
        smax[0] = M; smin[0] = m;
    }
    __syncthreads();
    float kmax = smax[0], kmin = smin[0];
    for (int l = t; l < L_; l += blockDim.x) {
        float q = Qs[l];
        float s = q * (q >= 0.f ? kmax : kmin);   // max_m(q*K_m) for stability
        float dsum = 0.f;
        for (int m = 0; m < L_; ++m) dsum += __expf(q * Ks[m] - s);
        Ss[l] = s;
        Rs[l] = 1.0f / ((float)L_ * dsum);
    }
    __syncthreads();
    for (int m = t; m < L_; m += blockDim.x) {
        float kv = Ks[m], acc = 0.f;
        for (int l = 0; l < L_; ++l) acc += __expf(Qs[l] * kv - Ss[l]) * Rs[l];
        imp[b * L_ + m] = acc;
    }
}

// ---------------- top-k via full in-LDS bitonic sort (descending, tie -> lower index) ----------------
__global__ void __launch_bounds__(1024) k_topk(const float* __restrict__ imp, int* __restrict__ topk) {
    __shared__ float v[L_];
    __shared__ int   id[L_];
    int b = blockIdx.x, t = threadIdx.x;
    for (int i = t; i < L_; i += 1024) { v[i] = imp[b * L_ + i]; id[i] = i; }
    __syncthreads();
    for (int size = 2; size <= L_; size <<= 1) {
        for (int stride = size >> 1; stride > 0; stride >>= 1) {
            __syncthreads();
#pragma unroll
            for (int q = 0; q < 2; ++q) {
                int i = t + (q << 10);
                int j = i ^ stride;
                if (j > i) {
                    bool desc = ((i & size) == 0);
                    float vi = v[i], vj = v[j];
                    int ii = id[i], ij = id[j];
                    bool iAfterJ = (vi < vj) || (vi == vj && ii > ij);
                    if (desc ? iAfterJ : !iAfterJ) {
                        v[i] = vj; v[j] = vi; id[i] = ij; id[j] = ii;
                    }
                }
            }
        }
    }
    __syncthreads();
    for (int i = t; i < KTOP_; i += 1024) topk[b * KTOP_ + i] = id[i];
}

// ---------------- fused gather + depthwise causal conv ----------------
__global__ void k_gconv(const float* __restrict__ xpn, const int* __restrict__ topk,
                        const float* __restrict__ cw, float* __restrict__ xc) {
    int gid = blockIdx.x * blockDim.x + threadIdx.x;
    const int total = B_ * KTOP_ * EXP_;
    if (gid >= total) return;
    int e = gid & (EXP_ - 1);
    int tmp = gid >> 9;            // / EXP_
    int tpos = tmp % KTOP_;
    int b = tmp / KTOP_;
    const int* ti = topk + b * KTOP_;
    float acc = 0.f;
#pragma unroll
    for (int d = 0; d < DCONV_; ++d) {
        int s = tpos - (DCONV_ - 1) + d;
        if (s >= 0) {
            int l = ti[s];
            acc += cw[e * DCONV_ + d] * xpn[((size_t)(b * L_ + l)) * EXP_ + e];
        }
    }
    xc[gid] = acc;
}

// ---------------- SSM scan: 2 channels/wave, state-per-lane, shuffle matvec ----------------
__global__ void k_scan(const float* __restrict__ xc, const float* __restrict__ Amat,
                       const float* __restrict__ Bp, const float* __restrict__ Cp,
                       float* __restrict__ outs) {
    int lane = threadIdx.x & 31;
    int wave = (blockIdx.x * blockDim.x + threadIdx.x) >> 5;
    int ch = wave * 2 + (lane >> 4);       // 0 .. B_*EXP_-1
    int m  = lane & 15;
    int b  = ch >> 9;                      // / EXP_
    int e  = ch & (EXP_ - 1);
    float a[NST_];
#pragma unroll
    for (int n = 0; n < NST_; ++n) a[n] = Amat[m * NST_ + n];
    float sb = 1.f / (1.f + __expf(-Bp[m]));              // sigB is per-state
    float sc = 1.f / (1.f + __expf(-Cp[e * NST_ + m]));
    float h = 0.f;
    int base = lane & 16;
    const float* xcp  = xc   + ((size_t)b * KTOP_) * EXP_ + e;
    float*       outp = outs + ((size_t)b * KTOP_) * EXP_ + e;
    for (int t = 0; t < KTOP_; ++t) {
        float xt = xcp[(size_t)t * EXP_];
        float hn = 0.f;
#pragma unroll
        for (int n = 0; n < NST_; ++n)
            hn += a[n] * __shfl(h, base + n, 32);
        h = hn + sb * xt;
        float p = h * sc;
#pragma unroll
        for (int off = 8; off >= 1; off >>= 1) p += __shfl_xor(p, off, 32);
        if (m == 0) outp[(size_t)t * EXP_] = p;
    }
}

// ---------------- residual copy ----------------
__global__ void k_copy(const float* __restrict__ x, float* __restrict__ out, int n) {
    int i = blockIdx.x * blockDim.x + threadIdx.x;
    if (i < n) out[i] = x[i];
}

// ---------------- scatter-add x_proc into out at topk positions ----------------
__global__ void k_scatter(const float* __restrict__ proj, const int* __restrict__ topk,
                          float* __restrict__ out) {
    int gid = blockIdx.x * blockDim.x + threadIdx.x;
    const int total = B_ * KTOP_ * D_;
    if (gid >= total) return;
    int d = gid & (D_ - 1);
    int tmp = gid >> 8;
    int tpos = tmp % KTOP_;
    int b = tmp / KTOP_;
    int l = topk[b * KTOP_ + tpos];
    out[((size_t)(b * L_ + l)) * D_ + d] += proj[gid];
}

extern "C" void kernel_launch(void* const* d_in, const int* in_sizes, int n_in,
                              void* d_out, int out_size, void* d_ws, size_t ws_size,
                              hipStream_t stream) {
    (void)in_sizes; (void)n_in; (void)out_size; (void)ws_size;
    const float* x     = (const float*)d_in[0];
    const float* alpha = (const float*)d_in[1];
    const float* dyt_w = (const float*)d_in[2];
    const float* dyt_b = (const float*)d_in[3];
    const float* w_in  = (const float*)d_in[4];
    const float* b_in  = (const float*)d_in[5];
    const float* w_q   = (const float*)d_in[6];
    const float* b_q   = (const float*)d_in[7];
    const float* w_k   = (const float*)d_in[8];
    const float* b_k   = (const float*)d_in[9];
    const float* Amat  = (const float*)d_in[10];
    const float* Bp    = (const float*)d_in[11];
    const float* Cp    = (const float*)d_in[12];
    const float* convw = (const float*)d_in[13];
    const float* w_out = (const float*)d_in[14];
    const float* b_out = (const float*)d_in[15];
    float* out = (float*)d_out;

    char* ws = (char*)d_ws;
    size_t off = 0;
    auto alloc = [&](size_t bytes) -> char* {
        char* p = ws + off;
        off = (off + bytes + 255) & ~(size_t)255;
        return p;
    };
    _Float16* w_in_h  = (_Float16*)alloc((size_t)EXP_ * D_ * 2);
    _Float16* w_out_h = (_Float16*)alloc((size_t)D_ * EXP_ * 2);
    _Float16* xn_h    = (_Float16*)alloc((size_t)B_ * L_ * D_ * 2);
    float*    xp      = (float*)   alloc((size_t)B_ * L_ * EXP_ * 4);   // normalized in place
    float*    Qv      = (float*)   alloc((size_t)B_ * L_ * 4);
    float*    Kv      = (float*)   alloc((size_t)B_ * L_ * 4);
    float*    imp     = (float*)   alloc((size_t)B_ * L_ * 4);
    int*      topk    = (int*)     alloc((size_t)B_ * KTOP_ * 4);
    float*    xc      = (float*)   alloc((size_t)B_ * KTOP_ * EXP_ * 4);
    float*    souts   = (float*)   alloc((size_t)B_ * KTOP_ * EXP_ * 4);
    float*    proj    = (float*)   alloc((size_t)B_ * KTOP_ * D_ * 4);

    const int nwin  = EXP_ * D_;          // 131072
    const int nxd   = B_ * L_ * D_;       // 4194304
    // 1) weight conversions
    k_cvt_f16<<<(nwin + 255) / 256, 256, 0, stream>>>(w_in,  w_in_h,  nwin);
    k_cvt_f16<<<(nwin + 255) / 256, 256, 0, stream>>>(w_out, w_out_h, nwin);
    // 2) DyT
    k_dyt<<<(nxd + 255) / 256, 256, 0, stream>>>(x, alpha, dyt_w, dyt_b, xn_h, nxd);
    // 3) in-projection GEMM (WMMA, 2x2 blocked): M=16384, N=512, K=256
    {
        int waves = ((B_ * L_ + 31) / 32) * (EXP_ / 32);   // 512 * 16 = 8192
        k_gemm2x2_f16<<<(waves + 7) / 8, 256, 0, stream>>>(xn_h, w_in_h, b_in, xp, B_ * L_, EXP_, D_);
    }
    // 4) row normalize + Q/K
    k_norm_qk<<<B_ * L_, 256, 0, stream>>>(xp, w_q, b_q, w_k, b_k, Qv, Kv);
    // 5) importance
    k_imp<<<B_, 1024, 0, stream>>>(Qv, Kv, imp);
    // 6) top-k (bitonic sort)
    k_topk<<<B_, 1024, 0, stream>>>(imp, topk);
    // 7) gather + depthwise causal conv
    {
        int total = B_ * KTOP_ * EXP_;
        k_gconv<<<(total + 255) / 256, 256, 0, stream>>>(xp, topk, convw, xc);
    }
    // 8) SSM scan: B_*EXP_ channels, 2 per wave -> 2048 waves -> 256 blocks of 256
    k_scan<<<(B_ * EXP_ * 16) / 256, 256, 0, stream>>>(xc, Amat, Bp, Cp, souts);
    // 9) out-projection GEMM (WMMA, 2x2 blocked, M-tail guarded): M=4912, N=256, K=512
    {
        int waves = ((B_ * KTOP_ + 31) / 32) * (D_ / 32);  // 154 * 8 = 1232
        k_gemm2x2_a32<<<(waves + 7) / 8, 256, 0, stream>>>(souts, w_out_h, b_out, proj, B_ * KTOP_, D_, EXP_);
    }
    // 10) residual copy then scatter-add
    k_copy<<<(nxd + 255) / 256, 256, 0, stream>>>(x, out, nxd);
    {
        int total = B_ * KTOP_ * D_;
        k_scatter<<<(total + 255) / 256, 256, 0, stream>>>(proj, topk, out);
    }
}
